// CausalMultiHeadSelfAttention_74723841016236
// MI455X (gfx1250) — matrix operand order repaired and checked by reading, not compiled, not run on hardware
//
#include <hip/hip_runtime.h>

typedef __attribute__((ext_vector_type(16))) _Float16 v16h;
typedef __attribute__((ext_vector_type(8)))  _Float16 v8h;
typedef __attribute__((ext_vector_type(8)))  float    v8f;
typedef __attribute__((ext_vector_type(4)))  unsigned int v4u;
typedef __attribute__((ext_vector_type(8)))  int      v8i_;
typedef __attribute__((ext_vector_type(4)))  int      v4i_;

#define D_MODEL 1024
#define SEQ     2048
#define NHEAD   16
#define DHEAD   64
#define BATCH   2
#define THETA_F 10000.0f

#if defined(__gfx1250__) && __has_builtin(__builtin_amdgcn_tensor_load_to_lds)
#define USE_TDM 1
#else
#define USE_TDM 0
#endif

// K-index inside a 16-bit A-fragment (16x32, MxK) for element e (0..15) of v16h.
// Contiguous order (matches B fills): lanes 0-15 -> K0..15, lanes 16-31 -> K16..31.
__device__ __forceinline__ int a_k_idx(int e, int laneHi) {
  return e + (laneHi ? 16 : 0);
}

__device__ __forceinline__ v8f wmma_f16(v16h a, v16h b, v8f c) {
  return __builtin_amdgcn_wmma_f32_16x16x32_f16(false, a, false, b, (short)0, c,
                                                false, false);
}

#if USE_TDM
// 2D tile DMA: global (row-major, row_stride elems) -> LDS, contiguous.
// dsize_code: 0=1B 1=2B 2=4B (ISA D# group1[17:16]).
__device__ __forceinline__ void tdm_load_2d(const void* gsrc, void* ldsdst,
                                            int dsize_code, int tile_w,
                                            int tile_h, int row_stride) {
  unsigned long long ga = (unsigned long long)(uintptr_t)gsrc;
  unsigned lds = (unsigned)(uintptr_t)ldsdst;  // low 32 bits = LDS byte offset
  v4u g0 = {1u,                                        // count=1 (valid user D#)
            lds,                                       // lds_addr
            (unsigned)(ga & 0xFFFFFFFFu),              // global_addr[31:0]
            (unsigned)((ga >> 32) & 0x01FFFFFFu) | (2u << 30)};  // [56:32]|type=2
  int td0 = row_stride;       // tensor_dim0 (elems)
  int td1 = 1 << 22;          // tensor_dim1: large (no OOB clamping)
  v8i_ g1 = {(dsize_code << 16),                              // data_size
             (td0 & 0xFFFF) << 16,                            // tensor_dim0 lo
             ((td0 >> 16) & 0xFFFF) | ((td1 & 0xFFFF) << 16), // dim0 hi | dim1 lo
             ((td1 >> 16) & 0xFFFF) | ((tile_w & 0xFFFF) << 16),  // | tile_dim0
             (tile_h & 0xFFFF),                               // tile_dim1, dim2=0
             td0,                                             // dim0_stride[31:0]
             0, 0};
  v4i_ gz = {0, 0, 0, 0};
#if __clang_major__ >= 23
  v8i_ gz8 = {0, 0, 0, 0, 0, 0, 0, 0};
  __builtin_amdgcn_tensor_load_to_lds(g0, g1, gz, gz, gz8, 0);
#else
  __builtin_amdgcn_tensor_load_to_lds(g0, g1, gz, gz, 0);
#endif
}
#endif

// ---------------------------------------------------------------------------
// Prep: vectorized f32 -> f16 conversion (done once; amortized over all tiles)
// ---------------------------------------------------------------------------
__global__ void cvt_f16_kernel(const float* __restrict__ src,
                               _Float16* __restrict__ dst, int n) {
  int i = (blockIdx.x * blockDim.x + threadIdx.x) * 8;
  if (i >= n) return;
  v8f x = *(const v8f*)(src + i);
  v8h h;
#pragma unroll
  for (int e = 0; e < 8; ++e) h[e] = (_Float16)x[e];
  *(v8h*)(dst + i) = h;
}

// ---------------------------------------------------------------------------
// Kernel A: O = Xh @ Wh^T (all-f16 WMMA, f32 acc), optional RoPE, f16 out.
// grid: ((B*S)/16, D/512, 3 {Q,K,V}), block 256; each wave: 16x64 (4 WMMA/K-step)
// ---------------------------------------------------------------------------
__global__ void qkv_rope_kernel(const _Float16* __restrict__ Xh,
                                const int* __restrict__ pos,
                                const _Float16* __restrict__ Wqh,
                                const _Float16* __restrict__ Wkh,
                                const _Float16* __restrict__ Wvh,
                                _Float16* __restrict__ Qf,
                                _Float16* __restrict__ Kf,
                                _Float16* __restrict__ Vf) {
  __shared__ _Float16 Ash[16][32];
  const int lane   = threadIdx.x & 31;
  const int wave   = threadIdx.x >> 5;
  const int laneHi = lane >> 4;
  const int n16    = lane & 15;
  const int rowBase = blockIdx.x * 16;           // row index in (B*S)
  const int which   = blockIdx.z;                // 0=Q 1=K 2=V
  const _Float16* W   = (which == 0) ? Wqh : ((which == 1) ? Wkh : Wvh);
  _Float16*       Out = (which == 0) ? Qf : ((which == 1) ? Kf : Vf);
  const int colBase = blockIdx.y * 512 + wave * 64;

  v8f acc[4] = {};
  for (int kb = 0; kb < D_MODEL / 32; ++kb) {
    const int K0 = kb * 32;
#if USE_TDM
    if (wave == 0) {  // TDM: one 16x32 f16 tile, Xh row-major stride D
      tdm_load_2d(Xh + (size_t)rowBase * D_MODEL + K0, &Ash[0][0],
                  /*2B*/ 1, 32, 16, D_MODEL);
      __builtin_amdgcn_s_wait_tensorcnt(0);
    }
#else
    {
      int idx = threadIdx.x * 2;
      int r = idx >> 5, c = idx & 31;
      const _Float16* src = Xh + (size_t)(rowBase + r) * D_MODEL + K0 + c;
      Ash[r][c]     = src[0];
      Ash[r][c + 1] = src[1];
    }
#endif
    __syncthreads();
    // Gather all operands first (independent regs -> overlapped loads),
    // then issue the 4 WMMAs back-to-back.
    v16h a, bfr[4];
#pragma unroll
    for (int e = 0; e < 16; ++e)
      a[e] = Ash[n16][a_k_idx(e, laneHi)];
#pragma unroll
    for (int t = 0; t < 4; ++t) {
#pragma unroll
      for (int e = 0; e < 16; ++e) {
        int kk = (laneHi ? 16 : 0) + e;  // B fragment: K across halves
        bfr[t][e] = W[(size_t)(colBase + t * 16 + n16) * D_MODEL + K0 + kk];
      }
    }
    __builtin_prefetch(&W[(size_t)(colBase + n16) * D_MODEL + K0 + 32], 0, 3);
#pragma unroll
    for (int t = 0; t < 4; ++t) acc[t] = wmma_f16(a, bfr[t], acc[t]);
    __syncthreads();
  }

#pragma unroll
  for (int t = 0; t < 4; ++t) {
#pragma unroll
    for (int r = 0; r < 8; ++r) {
      int m   = r + (laneHi ? 8 : 0);
      int row = rowBase + m;                // = b*S + s
      int s   = row & (SEQ - 1);
      int col = colBase + t * 16 + n16;
      float v = acc[t][r];
      float outv = v;
      if (which < 2) {                      // RoPE on Q and K
        float pv   = __shfl_xor(v, 1, 32);  // paired column in lane^1
        int dh     = col & (DHEAD - 1);
        int pairi  = dh >> 1;
        float p    = (float)pos[s];
        float invf =
            __expf(-((float)(2 * pairi) / (float)DHEAD) * __logf(THETA_F));
        float ang = p * invf;
        float c, sn;
        __sincosf(ang, &sn, &c);
        outv = ((col & 1) == 0) ? (v * c - pv * sn)   // even: xe*c - xo*s
                                : (pv * sn + v * c);  // odd:  xe*s + xo*c
      }
      Out[(size_t)row * D_MODEL + col] = (_Float16)outv;
    }
  }
}

// ---------------------------------------------------------------------------
// Kernel B: causal flash attention, one wave per 16 query rows per (b,h)
// grid: (S/128, H, B), block 256 (8 independent waves)
// ---------------------------------------------------------------------------
__global__ void flash_attn_kernel(const _Float16* __restrict__ Qf,
                                  const _Float16* __restrict__ Kf,
                                  const _Float16* __restrict__ Vf,
                                  _Float16* __restrict__ AO) {
  __shared__ _Float16 Plds[8][16][32];  // per-wave P transpose staging
  const int lane   = threadIdx.x & 31;
  const int wave   = threadIdx.x >> 5;
  const int laneHi = lane >> 4;
  const int n16    = lane & 15;
  const int qBase  = (blockIdx.x * 8 + wave) * 16;
  const int hOff   = blockIdx.y * DHEAD;
  const size_t rb  = (size_t)blockIdx.z * SEQ;

  v16h aq0, aq1;  // Q fragments for dh chunks [0,32) and [32,64)
#pragma unroll
  for (int e = 0; e < 16; ++e) {
    int k = a_k_idx(e, laneHi);
    size_t base = (rb + qBase + n16) * D_MODEL + hOff;
    aq0[e] = Qf[base + k];
    aq1[e] = Qf[base + 32 + k];
  }

  float rmax[8], rsum[8];
#pragma unroll
  for (int r = 0; r < 8; ++r) { rmax[r] = -3.0e38f; rsum[r] = 0.f; }
  v8f o0 = {}, o1 = {}, o2 = {}, o3 = {};

  const int nkb = (qBase + 16 + 31) / 32;  // causal: keys <= qBase+15
  for (int jb = 0; jb < nkb; ++jb) {
    const int keyBase = jb * 32;
    v8f s0 = {}, s1 = {};
    {
      v16h bk0, bk1, bk2, bk3;  // gather first, then 4 WMMAs
#pragma unroll
      for (int e = 0; e < 16; ++e) {
        int kk = (laneHi ? 16 : 0) + e;
        size_t c0 = (rb + keyBase + n16) * D_MODEL + hOff;
        size_t c1 = (rb + keyBase + 16 + n16) * D_MODEL + hOff;
        bk0[e] = Kf[c0 + kk];
        bk1[e] = Kf[c0 + 32 + kk];
        bk2[e] = Kf[c1 + kk];
        bk3[e] = Kf[c1 + 32 + kk];
      }
      s0 = wmma_f16(aq0, bk0, s0);
      s0 = wmma_f16(aq1, bk1, s0);
      s1 = wmma_f16(aq0, bk2, s1);
      s1 = wmma_f16(aq1, bk3, s1);
    }

    // mask + online softmax (row reductions across 16-lane halves)
#pragma unroll
    for (int r = 0; r < 8; ++r) {
      int m    = r + (laneHi ? 8 : 0);
      int q    = qBase + m;
      int key0 = keyBase + n16;
      float v0 = s0[r] * 0.125f;  // 1/sqrt(Dh)
      float v1 = s1[r] * 0.125f;
      if (key0 > q)      v0 = -3.0e38f;
      if (key0 + 16 > q) v1 = -3.0e38f;
      float mloc = fmaxf(v0, v1);
#pragma unroll
      for (int off = 8; off >= 1; off >>= 1)
        mloc = fmaxf(mloc, __shfl_xor(mloc, off, 32));
      float newmax = fmaxf(rmax[r], mloc);
      float scale  = __expf(rmax[r] - newmax);
      float p0 = __expf(v0 - newmax);
      float p1 = __expf(v1 - newmax);
      float ps = p0 + p1;
#pragma unroll
      for (int off = 8; off >= 1; off >>= 1)
        ps += __shfl_xor(ps, off, 32);
      rsum[r] = rsum[r] * scale + ps;
      rmax[r] = newmax;
      o0[r] *= scale; o1[r] *= scale; o2[r] *= scale; o3[r] *= scale;
      Plds[wave][m][n16]      = (_Float16)p0;  // C-layout -> A-layout transpose
      Plds[wave][m][16 + n16] = (_Float16)p1;
    }
    asm volatile("s_wait_dscnt 0" ::: "memory");

    v16h pf;
#pragma unroll
    for (int e = 0; e < 16; ++e)
      pf[e] = Plds[wave][n16][a_k_idx(e, laneHi)];

    v16h bv[4];
#pragma unroll
    for (int t = 0; t < 4; ++t) {
#pragma unroll
      for (int e = 0; e < 16; ++e) {
        int key = keyBase + (laneHi ? 16 : 0) + e;
        bv[t][e] = Vf[(rb + key) * D_MODEL + hOff + t * 16 + n16];
      }
    }
    o0 = wmma_f16(pf, bv[0], o0);
    o1 = wmma_f16(pf, bv[1], o1);
    o2 = wmma_f16(pf, bv[2], o2);
    o3 = wmma_f16(pf, bv[3], o3);
    asm volatile("s_wait_dscnt 0" ::: "memory");
  }

#pragma unroll
  for (int r = 0; r < 8; ++r) {
    int m = r + (laneHi ? 8 : 0);
    float inv = 1.f / rsum[r];
    size_t base = (rb + qBase + m) * D_MODEL + hOff + n16;
    AO[base]      = (_Float16)(o0[r] * inv);
    AO[base + 16] = (_Float16)(o1[r] * inv);
    AO[base + 32] = (_Float16)(o2[r] * inv);
    AO[base + 48] = (_Float16)(o3[r] * inv);
  }
}

// ---------------------------------------------------------------------------
// Kernel C: Out = AO @ Woh^T, f32 result. Same blocking as kernel A.
// ---------------------------------------------------------------------------
__global__ void out_proj_kernel(const _Float16* __restrict__ AO,
                                const _Float16* __restrict__ Woh,
                                float* __restrict__ Out) {
  __shared__ _Float16 Ash[16][32];
  const int lane   = threadIdx.x & 31;
  const int wave   = threadIdx.x >> 5;
  const int laneHi = lane >> 4;
  const int n16    = lane & 15;
  const int rowBase = blockIdx.x * 16;
  const int colBase = blockIdx.y * 512 + wave * 64;

  v8f acc[4] = {};
  for (int kb = 0; kb < D_MODEL / 32; ++kb) {
    const int K0 = kb * 32;
#if USE_TDM
    if (wave == 0) {  // TDM: one 16x32 f16 tile
      tdm_load_2d(AO + (size_t)rowBase * D_MODEL + K0, &Ash[0][0],
                  /*2B*/ 1, 32, 16, D_MODEL);
      __builtin_amdgcn_s_wait_tensorcnt(0);
    }
#else
    {
      int idx = threadIdx.x * 2;
      int r = idx >> 5, c = idx & 31;
      const _Float16* src = AO + (size_t)(rowBase + r) * D_MODEL + K0 + c;
      Ash[r][c]     = src[0];
      Ash[r][c + 1] = src[1];
    }
#endif
    __syncthreads();
    v16h a, bfr[4];
#pragma unroll
    for (int e = 0; e < 16; ++e)
      a[e] = Ash[n16][a_k_idx(e, laneHi)];
#pragma unroll
    for (int t = 0; t < 4; ++t) {
#pragma unroll
      for (int e = 0; e < 16; ++e) {
        int kk = (laneHi ? 16 : 0) + e;
        bfr[t][e] = Woh[(size_t)(colBase + t * 16 + n16) * D_MODEL + K0 + kk];
      }
    }
    __builtin_prefetch(&Woh[(size_t)(colBase + n16) * D_MODEL + K0 + 32], 0, 3);
#pragma unroll
    for (int t = 0; t < 4; ++t) acc[t] = wmma_f16(a, bfr[t], acc[t]);
    __syncthreads();
  }
#pragma unroll
  for (int t = 0; t < 4; ++t) {
#pragma unroll
    for (int r = 0; r < 8; ++r) {
      int m = r + (laneHi ? 8 : 0);
      Out[(size_t)(rowBase + m) * D_MODEL + colBase + t * 16 + n16] = acc[t][r];
    }
  }
}

// ---------------------------------------------------------------------------
extern "C" void kernel_launch(void* const* d_in, const int* in_sizes, int n_in,
                              void* d_out, int out_size, void* d_ws,
                              size_t ws_size, hipStream_t stream) {
  const float* X  = (const float*)d_in[0];
  const int*   tp = (const int*)d_in[1];
  const float* Wq = (const float*)d_in[2];
  const float* Wk = (const float*)d_in[3];
  const float* Wv = (const float*)d_in[4];
  const float* Wo = (const float*)d_in[5];
  float* Out = (float*)d_out;

  const size_t BSD = (size_t)BATCH * SEQ * D_MODEL;   // 4 Mi elems
  const size_t DD  = (size_t)D_MODEL * D_MODEL;       // 1 Mi elems
  _Float16* Xh  = (_Float16*)d_ws;
  _Float16* Qf  = Xh + BSD;
  _Float16* Kf  = Qf + BSD;
  _Float16* Vf  = Kf + BSD;
  _Float16* AO  = Vf + BSD;
  _Float16* Wqh = AO + BSD;
  _Float16* Wkh = Wqh + DD;
  _Float16* Wvh = Wkh + DD;
  _Float16* Woh = Wvh + DD;

  // one-time f32 -> f16 conversions (vectorized, 8 elems/thread)
  cvt_f16_kernel<<<(int)(BSD / 8 / 256), 256, 0, stream>>>(X, Xh, (int)BSD);
  cvt_f16_kernel<<<(int)(DD / 8 / 256), 256, 0, stream>>>(Wq, Wqh, (int)DD);
  cvt_f16_kernel<<<(int)(DD / 8 / 256), 256, 0, stream>>>(Wk, Wkh, (int)DD);
  cvt_f16_kernel<<<(int)(DD / 8 / 256), 256, 0, stream>>>(Wv, Wvh, (int)DD);
  cvt_f16_kernel<<<(int)(DD / 8 / 256), 256, 0, stream>>>(Wo, Woh, (int)DD);

  dim3 gA((BATCH * SEQ) / 16, D_MODEL / 512, 3);
  qkv_rope_kernel<<<gA, 256, 0, stream>>>(Xh, tp, Wqh, Wkh, Wvh, Qf, Kf, Vf);

  dim3 gB(SEQ / 128, NHEAD, BATCH);
  flash_attn_kernel<<<gB, 256, 0, stream>>>(Qf, Kf, Vf, AO);

  dim3 gC((BATCH * SEQ) / 16, D_MODEL / 512, 1);
  out_proj_kernel<<<gC, 256, 0, stream>>>(AO, Woh, Out);
}
